// KarrasOptimalDenoiser_48137993453822
// MI455X (gfx1250) — matrix operand order repaired and checked by reading, not compiled, not run on hardware
//
#include <hip/hip_runtime.h>
#include <hip/hip_bf16.h>

// ---------------------------------------------------------------------------
// Karras optimal denoiser:
//   logp = -0.5*||x-Y||^2/sigma^2 - D*log(sigma) - 0.5*D*log(2pi)
//   out  = softmax_n(logp) @ Y
// B=64, D=3072, N=50000.  HBM-bound (2 passes over 614MB of Y).
// Strategy: stream Y f32 -> bf16 LDS tiles -> v_wmma_f32_16x16x32_bf16.
// X tile (pre-converted bf16) staged via GLOBAL_LOAD_ASYNC_TO_LDS (ASYNCcnt).
// ---------------------------------------------------------------------------

typedef __attribute__((ext_vector_type(16))) __bf16 v16bf;
typedef __attribute__((ext_vector_type(8)))  float  v8f;
typedef __attribute__((ext_vector_type(4)))  int    v4i;

#define D_DIM 3072
#define LOG_2PI 1.8378770664093453f

// ---- gfx1250 async direct-to-LDS copy (guarded; falls back if absent) -----
#if __has_builtin(__builtin_amdgcn_global_load_async_to_lds_b128)
#define ASYNC_LDS_COPY 1
typedef __attribute__((address_space(1))) v4i gv4i_t;   // global int4
typedef __attribute__((address_space(3))) v4i lv4i_t;   // LDS int4
__device__ __forceinline__ void async_b128(const void* g, void* l) {
  gv4i_t* gp = (gv4i_t*)(unsigned long long)g;
  lv4i_t* lp = (lv4i_t*)(unsigned)(unsigned long long)l;  // low 32b = LDS offset
  __builtin_amdgcn_global_load_async_to_lds_b128(gp, lp, 0, 0);
}
__device__ __forceinline__ void async_wait0() {
#if __has_builtin(__builtin_amdgcn_s_wait_asynccnt)
  __builtin_amdgcn_s_wait_asynccnt(0);
#else
  asm volatile("s_wait_asynccnt 0x0" ::: "memory");
#endif
}
#endif

__device__ __forceinline__ unsigned short f2bf(float f) {
  // round-to-nearest-even f32 -> bf16
  unsigned int u = __float_as_uint(f);
  u += 0x7FFFu + ((u >> 16) & 1u);
  return (unsigned short)(u >> 16);
}

// A fragment (16x32 bf16, MxK). Element (m,k) lives at base[m*ld + k].
// Layout (ISA 7.12.2): lanes0-15: M=lane, dwords0-3=K0..7, dwords4-7=K16..23;
// lanes16-31: M=lane-16, dwords0-3=K8..15, dwords4-7=K24..31.
__device__ __forceinline__ v16bf load_a_frag(const unsigned short* base, int ld,
                                             int row0, int koff, int lane) {
  union { v16bf v; unsigned int u[8]; } f;
  const int r = row0 + (lane & 15);
  const int kadd = (lane >> 4) * 8;
  const unsigned short* p = base + r * ld + koff + kadd;
#pragma unroll
  for (int j = 0; j < 4; ++j) f.u[j]     = *(const unsigned int*)(p + 2 * j);
#pragma unroll
  for (int j = 0; j < 4; ++j) f.u[4 + j] = *(const unsigned int*)(p + 16 + 2 * j);
  return f.v;
}

// B fragment (32x16 bf16, KxN). Element (k,n) lives at base[n*ld + k]
// (K contiguous per column -> one dword = 2 consecutive K).
// Layout: lanes0-15: N=lane, K=koff+0..7,16..23; lanes16-31: N=lane-16, K=koff+8..15,24..31.
__device__ __forceinline__ v16bf load_b_frag(const unsigned short* base, int ld,
                                             int col0, int koff, int lane) {
  union { v16bf v; unsigned int u[8]; } f;
  const int c = col0 + (lane & 15);
  const int kb = koff + (lane >> 4) * 8;
  const unsigned short* p = base + c * ld + kb;
#pragma unroll
  for (int j = 0; j < 4; ++j) f.u[j] = *(const unsigned int*)(p + 2 * j);
#pragma unroll
  for (int j = 0; j < 4; ++j) f.u[4 + j] = *(const unsigned int*)(p + 16 + 2 * j);
  return f.v;
}

__device__ __forceinline__ v8f wmma_bf16(v16bf a, v16bf b, v8f c) {
  return __builtin_amdgcn_wmma_f32_16x16x32_bf16(false, a, false, b,
                                                 (short)0, c, false, false);
}

// ---------------------------------------------------------------------------
// Kernel 0: x -> bf16, x2 = ||x_b||^2, zero d_out.  grid=64 blocks (one per b)
// ---------------------------------------------------------------------------
__global__ __launch_bounds__(256) void k_prep(const float* __restrict__ x,
                                              float* __restrict__ x2,
                                              unsigned short* __restrict__ xb,
                                              float* __restrict__ outz) {
  const int b = blockIdx.x, t = threadIdx.x;
  float s = 0.f;
  for (int i = t; i < D_DIM; i += 256) {
    float v = x[(size_t)b * D_DIM + i];
    s += v * v;
    xb[(size_t)b * D_DIM + i] = f2bf(v);
    outz[(size_t)b * D_DIM + i] = 0.f;
  }
#pragma unroll
  for (int o = 16; o; o >>= 1) s += __shfl_xor(s, o, 32);
  __shared__ float red[8];
  const int lane = t & 31, wid = t >> 5;
  if (lane == 0) red[wid] = s;
  __syncthreads();
  if (t == 0) {
    float tot = 0.f;
#pragma unroll
    for (int i = 0; i < 8; ++i) tot += red[i];
    x2[b] = tot;
  }
}

// ---------------------------------------------------------------------------
// Kernel 1: logp[64 x Npad] tile GEMM.  One block = 128 Y rows, K=3072.
// Fuses ||y||^2 and the log-normal epilogue into the single pass over Y.
// ---------------------------------------------------------------------------
__global__ __launch_bounds__(256) void k_logp(const float* __restrict__ Y,
                                              const float* __restrict__ sigma,
                                              const unsigned short* __restrict__ xb,
                                              const float* __restrict__ x2,
                                              float* __restrict__ logp,
                                              int N, int Npad) {
  __shared__ unsigned short Xt[64][64];    // bf16 X tile  (M x K)
  __shared__ unsigned short Yt[128][64];   // bf16 Y tile  (N x K), K contiguous
  __shared__ float y2s[128];
  __shared__ float x2s[64], invv[64], sterm[64];

  const int t = threadIdx.x;
  const int lane = t & 31, wid = t >> 5;
  const int n0 = blockIdx.x * 128;

  if (t < 64) {
    float sg = sigma[t];
    x2s[t] = x2[t];
    invv[t] = 1.0f / (sg * sg);
    sterm[t] = -(float)D_DIM * __logf(sg) - 0.5f * (float)D_DIM * LOG_2PI;
  }

  // each wave owns a 16(M) x 64(N) patch: mstrip in 0..3, col base 0 or 64
  const int mstrip = wid & 3;
  const int cbase = (wid >> 2) * 64;
  v8f acc0 = {}, acc1 = {}, acc2 = {}, acc3 = {};

  // Y streaming assignment: 2 threads per row, 32 cols each
  const int yrow = t >> 1;
  const int khalf = (t & 1) * 32;
  const int gn = n0 + yrow;
  const bool nvalid = gn < N;
  float y2p = 0.f;

  // X staging assignment
  const int xr = t >> 2, seg = (t & 3) * 16;
  const unsigned short* xsrc = xb + (size_t)xr * D_DIM + seg;

  for (int k0 = 0; k0 < D_DIM; k0 += 64) {
    // --- stage X tile (already bf16 in ws): async direct-to-LDS if available
#ifdef ASYNC_LDS_COPY
    async_b128(xsrc + k0,     &Xt[xr][seg]);
    async_b128(xsrc + k0 + 8, &Xt[xr][seg + 8]);
#else
    {
      const uint4* src = (const uint4*)(xsrc + k0);
      uint4 a = src[0], b = src[1];
      *(uint4*)&Xt[xr][seg] = a;
      *(uint4*)&Xt[xr][seg + 8] = b;
    }
#endif
    // --- stage Y tile: f32 -> bf16, fused ||y||^2 partial ---
    {
      const float* src = Y + (size_t)gn * D_DIM + k0 + khalf;
      if (nvalid && k0 + 64 < D_DIM)
        __builtin_prefetch(src + 64, 0, 1);  // global_prefetch next k-tile
#pragma unroll
      for (int j = 0; j < 8; ++j) {
        float4 v = nvalid ? ((const float4*)src)[j] : make_float4(0.f, 0.f, 0.f, 0.f);
        y2p += v.x * v.x + v.y * v.y + v.z * v.z + v.w * v.w;
        Yt[yrow][khalf + j * 4 + 0] = f2bf(v.x);
        Yt[yrow][khalf + j * 4 + 1] = f2bf(v.y);
        Yt[yrow][khalf + j * 4 + 2] = f2bf(v.z);
        Yt[yrow][khalf + j * 4 + 3] = f2bf(v.w);
      }
    }
#ifdef ASYNC_LDS_COPY
    async_wait0();   // ASYNCcnt==0: X tile landed in LDS (overlapped Y staging)
#endif
    __syncthreads();
#pragma unroll
    for (int s = 0; s < 2; ++s) {
      const int koff = s * 32;
      v16bf A  = load_a_frag(&Xt[0][0], 64, mstrip * 16, koff, lane);
      v16bf B0 = load_b_frag(&Yt[0][0], 64, cbase + 0,  koff, lane);
      acc0 = wmma_bf16(A, B0, acc0);
      v16bf B1 = load_b_frag(&Yt[0][0], 64, cbase + 16, koff, lane);
      acc1 = wmma_bf16(A, B1, acc1);
      v16bf B2 = load_b_frag(&Yt[0][0], 64, cbase + 32, koff, lane);
      acc2 = wmma_bf16(A, B2, acc2);
      v16bf B3 = load_b_frag(&Yt[0][0], 64, cbase + 48, koff, lane);
      acc3 = wmma_bf16(A, B3, acc3);
    }
    __syncthreads();
  }

  // finalize ||y||^2: pair (t, t^1) are adjacent lanes of the same wave
  {
    float tot = y2p + __shfl_xor(y2p, 1, 32);
    if ((t & 1) == 0) y2s[yrow] = tot;
  }
  __syncthreads();

  // epilogue: C layout -> dword r: lanes0-15 M=r, lanes16-31 M=8+r; N=lane&15
  const int mrow_off = (lane >> 4) * 8;
  const int ncol = lane & 15;
  auto store_tile = [&](v8f a, int ct) {
    const int n_local = cbase + ct * 16 + ncol;
    const int n = n0 + n_local;
    const float y2v = y2s[n_local];
#pragma unroll
    for (int r = 0; r < 8; ++r) {
      const int b = mstrip * 16 + mrow_off + r;
      float d2 = x2s[b] + y2v - 2.0f * a[r];
      d2 = fmaxf(d2, 0.0f);
      float val = -0.5f * d2 * invv[b] + sterm[b];
      if (n >= N) val = -1e30f;  // padded rows get zero softmax weight
      logp[(size_t)b * Npad + n] = val;
    }
  };
  store_tile(acc0, 0); store_tile(acc1, 1); store_tile(acc2, 2); store_tile(acc3, 3);
}

// ---------------------------------------------------------------------------
// Kernel 2: per-row online logsumexp -> mrow[b], lrow[b].  grid=64
// ---------------------------------------------------------------------------
__global__ __launch_bounds__(256) void k_stats(const float* __restrict__ logp,
                                               float* __restrict__ mrow,
                                               float* __restrict__ lrow, int Npad) {
  const int b = blockIdx.x, t = threadIdx.x;
  const int lane = t & 31, wid = t >> 5;
  const float* row = logp + (size_t)b * Npad;
  float m = -3.0e38f, s = 0.f;
  for (int i = t; i < Npad; i += 256) {
    float v = row[i];
    float nm = fmaxf(m, v);
    s = s * __expf(m - nm) + __expf(v - nm);
    m = nm;
  }
#pragma unroll
  for (int o = 16; o; o >>= 1) {
    float m2 = __shfl_xor(m, o, 32);
    float s2 = __shfl_xor(s, o, 32);
    float nm = fmaxf(m, m2);
    s = s * __expf(m - nm) + s2 * __expf(m2 - nm);
    m = nm;
  }
  __shared__ float sm[8], ss[8];
  if (lane == 0) { sm[wid] = m; ss[wid] = s; }
  __syncthreads();
  if (t == 0) {
    float M = sm[0], S = ss[0];
#pragma unroll
    for (int i = 1; i < 8; ++i) {
      float nm = fmaxf(M, sm[i]);
      S = S * __expf(M - nm) + ss[i] * __expf(sm[i] - nm);
      M = nm;
    }
    mrow[b] = M;
    lrow[b] = S;
  }
}

// ---------------------------------------------------------------------------
// Kernel 3: out[64 x 3072] = w @ Y, K = n (split over grid.y, f32 atomics).
// w = exp(logp - m)/l computed on the fly, staged bf16.
// ---------------------------------------------------------------------------
__global__ __launch_bounds__(256) void k_out(const float* __restrict__ Y,
                                             const float* __restrict__ logp,
                                             const float* __restrict__ mrow,
                                             const float* __restrict__ lrow,
                                             float* __restrict__ out,
                                             int N, int Npad) {
  __shared__ unsigned short Wt[64][32];   // A: (b x n) bf16, K contiguous
  __shared__ unsigned short Yt[128][32];  // B: [col][k] bf16, K contiguous

  const int t = threadIdx.x, lane = t & 31, wid = t >> 5;
  const int c0 = blockIdx.x * 128;  // output column base
  const int nChunks = Npad / 32;
  const int per = (nChunks + gridDim.y - 1) / gridDim.y;
  const int kcBeg = blockIdx.y * per;
  const int kcEnd = (kcBeg + per < nChunks) ? (kcBeg + per) : nChunks;

  // W-fill assignment (b fixed per thread -> hoist row stats)
  const int wb = t >> 2;
  const int wn = (t & 3) * 8;
  const float mb = mrow[wb];
  const float invl = 1.0f / lrow[wb];

  // Y-fill assignment: one k-row per 8 threads, 16 cols each
  const int yr = t >> 3;
  const int yc = (t & 7) * 16;

  const int mstrip = wid & 3;
  const int cbase = (wid >> 2) * 64;
  v8f acc0 = {}, acc1 = {}, acc2 = {}, acc3 = {};

  for (int kc = kcBeg; kc < kcEnd; ++kc) {
    const int nbase = kc * 32;
    // stage softmax weights
#pragma unroll
    for (int i = 0; i < 8; ++i) {
      const int n = nbase + wn + i;
      float lp = logp[(size_t)wb * Npad + n];
      Wt[wb][wn + i] = f2bf(__expf(lp - mb) * invl);
    }
    // stage Y tile transposed ([col][k]) with f32->bf16
    {
      const int n = nbase + yr;
      const bool v = n < N;
      const float4* src = (const float4*)(Y + (size_t)n * D_DIM + c0 + yc);
      if (v && kc + 1 < kcEnd)
        __builtin_prefetch((const float*)src + 32 * D_DIM, 0, 1);
#pragma unroll
      for (int j = 0; j < 4; ++j) {
        float4 q = v ? src[j] : make_float4(0.f, 0.f, 0.f, 0.f);
        Yt[yc + j * 4 + 0][yr] = f2bf(q.x);
        Yt[yc + j * 4 + 1][yr] = f2bf(q.y);
        Yt[yc + j * 4 + 2][yr] = f2bf(q.z);
        Yt[yc + j * 4 + 3][yr] = f2bf(q.w);
      }
    }
    __syncthreads();
    v16bf A  = load_a_frag(&Wt[0][0], 32, mstrip * 16, 0, lane);
    v16bf B0 = load_b_frag(&Yt[0][0], 32, cbase + 0,  0, lane);
    acc0 = wmma_bf16(A, B0, acc0);
    v16bf B1 = load_b_frag(&Yt[0][0], 32, cbase + 16, 0, lane);
    acc1 = wmma_bf16(A, B1, acc1);
    v16bf B2 = load_b_frag(&Yt[0][0], 32, cbase + 32, 0, lane);
    acc2 = wmma_bf16(A, B2, acc2);
    v16bf B3 = load_b_frag(&Yt[0][0], 32, cbase + 48, 0, lane);
    acc3 = wmma_bf16(A, B3, acc3);
    __syncthreads();
  }

  const int mrow_off = (lane >> 4) * 8;
  const int ncol = lane & 15;
  auto st = [&](v8f a, int ct) {
    const int c = c0 + cbase + ct * 16 + ncol;
#pragma unroll
    for (int r = 0; r < 8; ++r) {
      const int b = mstrip * 16 + mrow_off + r;
      atomicAdd(&out[(size_t)b * D_DIM + c], a[r]);
    }
  };
  st(acc0, 0); st(acc1, 1); st(acc2, 2); st(acc3, 3);
}

// ---------------------------------------------------------------------------
extern "C" void kernel_launch(void* const* d_in, const int* in_sizes, int n_in,
                              void* d_out, int out_size, void* d_ws, size_t ws_size,
                              hipStream_t stream) {
  (void)n_in; (void)out_size; (void)ws_size;
  const float* x     = (const float*)d_in[0];  // [64,3,32,32]
  const float* sigma = (const float*)d_in[1];  // [64]
  const float* Y     = (const float*)d_in[2];  // [N,3,32,32]
  float* out = (float*)d_out;                  // [64,3,32,32]

  const int N = in_sizes[2] / D_DIM;           // 50000
  const int Npad = ((N + 127) / 128) * 128;    // 50048

  // workspace layout
  float* ws = (float*)d_ws;
  float* x2   = ws;                                     // 64
  float* mrow = ws + 64;                                // 64
  float* lrow = ws + 128;                               // 64
  unsigned short* xb = (unsigned short*)(ws + 192);     // 64*3072 bf16
  float* logp = (float*)(xb + (size_t)64 * D_DIM);      // 64*Npad f32

  k_prep<<<64, 256, 0, stream>>>(x, x2, xb, out);
  k_logp<<<Npad / 128, 256, 0, stream>>>(Y, sigma, xb, x2, logp, N, Npad);
  k_stats<<<64, 256, 0, stream>>>(logp, mrow, lrow, Npad);
  dim3 g3(D_DIM / 128, 8);
  k_out<<<g3, 256, 0, stream>>>(Y, logp, mrow, lrow, out, N, Npad);
}
